// GraphFusion_62328565399968
// MI455X (gfx1250) — matrix operand order repaired and checked by reading, not compiled, not run on hardware
//
#include <hip/hip_runtime.h>

#define BATCH  64
#define TT     512
#define LL     4
#define II     4
#define HH     768
#define NNODE  520          // TT + LL + II
#define NHEAD  4
#define DHEAD  192
#define NLAYER 3
#define MT     33           // ceil(520/16) M-tiles
#define LN_EPS 1e-5f
#define NEG_INF (-3.0e38f)

static_assert(NHEAD * DHEAD == HH, "head split");

typedef __attribute__((ext_vector_type(16))) __bf16       v16bf;
typedef __attribute__((ext_vector_type(8)))  float        v8f;
typedef __attribute__((ext_vector_type(8)))  unsigned int v8u;

// round-to-nearest-even fp32 -> bf16, packed pair into one dword
__device__ __forceinline__ unsigned int pack2bf(float a, float b) {
  unsigned int ua = __builtin_bit_cast(unsigned int, a);
  unsigned int ub = __builtin_bit_cast(unsigned int, b);
  ua = (ua + 0x7FFFu + ((ua >> 16) & 1u)) >> 16;
  ub = (ub + 0x7FFFu + ((ub >> 16) & 1u)) >> 16;
  return ua | (ub << 16);
}

// ---------------------------------------------------------------- build x = concat(text,label,image)
__global__ void build_x_kernel(const float* __restrict__ t, const float* __restrict__ l,
                               const float* __restrict__ im, float* __restrict__ x, long total) {
  long idx = (long)blockIdx.x * blockDim.x + threadIdx.x;
  if (idx >= total) return;
  int c = (int)(idx % HH);
  long r = idx / HH;
  int node = (int)(r % NNODE);
  int b = (int)(r / NNODE);
  float v;
  if (node < TT)            v = t [((long)b * TT + node) * HH + c];
  else if (node < TT + LL)  v = l [((long)b * LL + (node - TT)) * HH + c];
  else                      v = im[((long)b * II + (node - TT - LL)) * HH + c];
  x[idx] = v;
}

// ---------------------------------------------------------------- pack W into WMMA B-layout (bf16)
// B-tile (32K x 16N) for (ntile,kstep): lane L holds N = n0+(L&15); elems 0..7 = K g*8..g*8+7, elems 8..15 = +16 (g = L>>4).
__global__ void pack_w_kernel(const float* __restrict__ Ws, unsigned int* __restrict__ Wp) {
  int ks = blockIdx.x;   // 0..23
  int nt = blockIdx.y;   // 0..47
  int l  = blockIdx.z;   // 0..NLAYER-1
  int lane = threadIdx.x;
  const float* W = Ws + (size_t)l * HH * HH;
  unsigned int* o = Wp + ((((size_t)l * 48 + nt) * 24 + ks) * 32 + lane) * 8;
  int g  = lane >> 4;
  int n  = nt * 16 + (lane & 15);
#pragma unroll
  for (int d = 0; d < 8; ++d) {
    // B[k][n] = W[n][k]  (h = x @ W.T)
    o[d] = pack2bf(W[(size_t)n * HH + ks * 32 + (d < 4 ? 8 * g + 2 * d : 8 + 8 * g + 2 * d)],
                   W[(size_t)n * HH + ks * 32 + (d < 4 ? 8 * g + 2 * d : 8 + 8 * g + 2 * d) + 1]);
  }
}

// ---------------------------------------------------------------- pack x into WMMA A-layout (bf16)
// A-tile (16M x 32K): lane = M (lane&15); elems 0..7 = K g*8..g*8+7, elems 8..15 = +16 (g = lane>>4)
__global__ void pack_a_kernel(const float* __restrict__ X, unsigned int* __restrict__ Ap) {
  const int ks   = blockIdx.x;   // 0..23
  const int mt   = blockIdx.y;   // 0..MT-1
  const int b    = blockIdx.z;
  const int lane = threadIdx.x;
  const int g = lane >> 4, m = lane & 15;
  const int row = mt * 16 + m;
  unsigned int* o = Ap + (((size_t)(b * MT + mt) * 24 + ks) * 32 + lane) * 8;
  if (row < NNODE) {
    const float* xrow = X + ((size_t)b * NNODE + row) * HH;
    const int kk = ks * 32 + g * 8;
    v8f lo = *(const v8f*)(xrow + kk);        // 32B aligned
    v8f hi = *(const v8f*)(xrow + kk + 16);
#pragma unroll
    for (int r = 0; r < 4; ++r) {
      o[r]     = pack2bf(lo[2 * r], lo[2 * r + 1]);
      o[4 + r] = pack2bf(hi[2 * r], hi[2 * r + 1]);
    }
  } else {
#pragma unroll
    for (int d = 0; d < 8; ++d) o[d] = 0u;
  }
}

// ---------------------------------------------------------------- GEMM: h = x @ W.T via bf16 WMMA
// One wave: 16 rows x 64 cols output, K=768 in 24 steps of 32. A and B both pre-packed:
// inner loop is pure b128 loads + 4x v_wmma per step.
__global__ void gemm_wmma_kernel(const unsigned int* __restrict__ Ap,
                                 const unsigned int* __restrict__ Wp,   // layer-offset base
                                 float* __restrict__ Hout) {
  const int lane = threadIdx.x;        // 0..31
  const int quad = blockIdx.x;         // 0..11  (4 N-tiles each)
  const int mt   = blockIdx.y;         // 0..MT-1
  const int b    = blockIdx.z;
  const int g    = lane >> 4;
  const int m    = lane & 15;

  v8f acc[4];
#pragma unroll
  for (int j = 0; j < 4; ++j) acc[j] = (v8f){0, 0, 0, 0, 0, 0, 0, 0};

  const v8u* ap = (const v8u*)Ap + ((size_t)(b * MT + mt) * 24 * 32 + lane);
  const v8u* bp0 = (const v8u*)Wp + ((size_t)(quad * 4 + 0) * 24 * 32 + lane);
  const v8u* bp1 = (const v8u*)Wp + ((size_t)(quad * 4 + 1) * 24 * 32 + lane);
  const v8u* bp2 = (const v8u*)Wp + ((size_t)(quad * 4 + 2) * 24 * 32 + lane);
  const v8u* bp3 = (const v8u*)Wp + ((size_t)(quad * 4 + 3) * 24 * 32 + lane);

  for (int ks = 0; ks < 24; ++ks) {
    const int o = ks * 32;                       // v8u stride per k-step
    v16bf a  = __builtin_bit_cast(v16bf, ap[o]);
    v16bf b0 = __builtin_bit_cast(v16bf, bp0[o]);
    v16bf b1 = __builtin_bit_cast(v16bf, bp1[o]);
    v16bf b2 = __builtin_bit_cast(v16bf, bp2[o]);
    v16bf b3 = __builtin_bit_cast(v16bf, bp3[o]);
    acc[0] = __builtin_amdgcn_wmma_f32_16x16x32_bf16(false, a, false, b0, (short)0, acc[0], false, false);
    acc[1] = __builtin_amdgcn_wmma_f32_16x16x32_bf16(false, a, false, b1, (short)0, acc[1], false, false);
    acc[2] = __builtin_amdgcn_wmma_f32_16x16x32_bf16(false, a, false, b2, (short)0, acc[2], false, false);
    acc[3] = __builtin_amdgcn_wmma_f32_16x16x32_bf16(false, a, false, b3, (short)0, acc[3], false, false);
  }

  // C/D layout: VGPR v -> row mt*16 + g*8 + v, col = n0 + (lane&15)
  float* hbase = Hout + (size_t)b * NNODE * HH;
#pragma unroll
  for (int j = 0; j < 4; ++j) {
    const int col = (quad * 4 + j) * 16 + m;
#pragma unroll
    for (int v = 0; v < 8; ++v) {
      const int r2 = mt * 16 + g * 8 + v;
      if (r2 < NNODE) hbase[(size_t)r2 * HH + col] = acc[j][v];
    }
  }
}

// ---------------------------------------------------------------- top-k(3) cosine edges per text node
__global__ void topk_kernel(const float* __restrict__ text, const float* __restrict__ label,
                            const float* __restrict__ image,
                            int* __restrict__ idxl, int* __restrict__ idxi) {
  const int t = blockIdx.x, b = blockIdx.y, tid = threadIdx.x;   // block 128
  const float* tr = text  + ((size_t)b * TT + t) * HH;
  const float* lb = label + (size_t)b * LL * HH;
  const float* im = image + (size_t)b * II * HH;
  float dl[4] = {0, 0, 0, 0}, di[4] = {0, 0, 0, 0};
  float nl[4] = {0, 0, 0, 0}, ni[4] = {0, 0, 0, 0}, tn = 0.f;
  for (int c = tid; c < HH; c += 128) {
    float tv = tr[c]; tn += tv * tv;
#pragma unroll
    for (int j = 0; j < 4; ++j) {
      float lv = lb[(size_t)j * HH + c]; dl[j] += tv * lv; nl[j] += lv * lv;
      float iv = im[(size_t)j * HH + c]; di[j] += tv * iv; ni[j] += iv * iv;
    }
  }
  __shared__ float sh[17][128];
  sh[0][tid] = tn;
#pragma unroll
  for (int j = 0; j < 4; ++j) {
    sh[1 + j][tid]  = dl[j]; sh[5 + j][tid]  = di[j];
    sh[9 + j][tid]  = nl[j]; sh[13 + j][tid] = ni[j];
  }
  __syncthreads();
  for (int s = 64; s > 0; s >>= 1) {
    if (tid < s)
      for (int q = 0; q < 17; ++q) sh[q][tid] += sh[q][tid + s];
    __syncthreads();
  }
  if (tid == 0) {
    float tnorm = fmaxf(sqrtf(sh[0][0]), 1e-8f);
    float cl[4], ci[4];
#pragma unroll
    for (int j = 0; j < 4; ++j) {
      cl[j] = sh[1 + j][0] / (tnorm * fmaxf(sqrtf(sh[9 + j][0]),  1e-8f));
      ci[j] = sh[5 + j][0] / (tnorm * fmaxf(sqrtf(sh[13 + j][0]), 1e-8f));
    }
    int* ol = idxl + ((size_t)b * TT + t) * 3;
    int* oi = idxi + ((size_t)b * TT + t) * 3;
    int used = 0;
    for (int r = 0; r < 3; ++r) {                 // stable top-3 (ties -> lower index)
      int best = 0; float bv = NEG_INF;
      for (int j = 0; j < 4; ++j)
        if (!((used >> j) & 1) && cl[j] > bv) { bv = cl[j]; best = j; }
      used |= 1 << best; ol[r] = best;
    }
    used = 0;
    for (int r = 0; r < 3; ++r) {
      int best = 0; float bv = NEG_INF;
      for (int j = 0; j < 4; ++j)
        if (!((used >> j) & 1) && ci[j] > bv) { bv = ci[j]; best = j; }
      used |= 1 << best; oi[r] = best;
    }
  }
}

// ---------------------------------------------------------------- per-node attention scores s_src/s_dst
__global__ void scores_kernel(const float* __restrict__ Hmat, const float* __restrict__ a_s,
                              const float* __restrict__ a_d,
                              float* __restrict__ ssrc, float* __restrict__ sdst) {
  const int node = blockIdx.x, b = blockIdx.y, tid = threadIdx.x;  // block 256, 3 ch/thread
  const float* hr = Hmat + ((size_t)b * NNODE + node) * HH;
  float ps = 0.f, pd = 0.f;
#pragma unroll
  for (int q = 0; q < 3; ++q) {
    int c = tid * 3 + q;
    float hv = hr[c];
    ps += hv * a_s[c]; pd += hv * a_d[c];
  }
  __shared__ float shs[256], shd[256];
  shs[tid] = ps; shd[tid] = pd; __syncthreads();
  for (int s = 32; s > 0; s >>= 1) {                 // reduce within each 64-thread (head) group
    if ((tid & 63) < s) { shs[tid] += shs[tid + s]; shd[tid] += shd[tid + s]; }
    __syncthreads();
  }
  if ((tid & 63) == 0) {
    int hd = tid >> 6;
    ssrc[((size_t)b * NNODE + node) * NHEAD + hd] = shs[tid];
    sdst[((size_t)b * NNODE + node) * NHEAD + hd] = shd[tid];
  }
}

// ---------------------------------------------------------------- text-dst nodes: fused GAT + ReLU + residual + LN
__global__ void agg_text_kernel(const float* __restrict__ Hmat, const float* __restrict__ ssrc,
                                const float* __restrict__ sdst, const int* __restrict__ idxl,
                                const int* __restrict__ idxi, const float* __restrict__ bias,
                                const float* __restrict__ gamma, const float* __restrict__ beta,
                                float* __restrict__ x, float* __restrict__ outp) {
  const int t = blockIdx.x, b = blockIdx.y, tid = threadIdx.x;   // block 256, 3 ch/thread
  int srcs[9]; int ns = 0;
  if (t > 0)      srcs[ns++] = t - 1;
  if (t < TT - 1) srcs[ns++] = t + 1;
  srcs[ns++] = t;                                   // self loop
  const int* il = idxl + ((size_t)b * TT + t) * 3;
  const int* ii = idxi + ((size_t)b * TT + t) * 3;
  for (int r = 0; r < 3; ++r) srcs[ns++] = TT + il[r];
  for (int r = 0; r < 3; ++r) srcs[ns++] = TT + LL + ii[r];

  const int hd = tid >> 6;
  const float sd = sdst[((size_t)b * NNODE + t) * NHEAD + hd];
  float e[9]; float m = NEG_INF;
  for (int j = 0; j < ns; ++j) {
    float v = ssrc[((size_t)b * NNODE + srcs[j]) * NHEAD + hd] + sd;
    v = v > 0.f ? v : 0.2f * v;                     // LeakyReLU(0.2)
    e[j] = v; m = fmaxf(m, v);
  }
  float den = 0.f;
  for (int j = 0; j < ns; ++j) { e[j] = __expf(e[j] - m); den += e[j]; }
  const float inv = 1.0f / (den + 1e-16f);

  float acc[3] = {0.f, 0.f, 0.f};
  for (int j = 0; j < ns; ++j) {
    const float* hr = Hmat + ((size_t)b * NNODE + srcs[j]) * HH;
    const float w = e[j] * inv;
#pragma unroll
    for (int q = 0; q < 3; ++q) acc[q] += w * hr[tid * 3 + q];
  }

  float* xr = x + ((size_t)b * NNODE + t) * HH;
  float y[3]; float s1 = 0.f, s2 = 0.f;
#pragma unroll
  for (int q = 0; q < 3; ++q) {
    int c = tid * 3 + q;
    float o = fmaxf(acc[q] + bias[c], 0.f);
    float yy = o + xr[c];
    y[q] = yy; s1 += yy; s2 += yy * yy;
  }
  __shared__ float r1[256], r2[256];
  r1[tid] = s1; r2[tid] = s2; __syncthreads();
  for (int s = 128; s > 0; s >>= 1) {
    if (tid < s) { r1[tid] += r1[tid + s]; r2[tid] += r2[tid + s]; }
    __syncthreads();
  }
  const float mu = r1[0] * (1.0f / HH);
  const float var = r2[0] * (1.0f / HH) - mu * mu;
  const float rstd = rsqrtf(var + LN_EPS);
#pragma unroll
  for (int q = 0; q < 3; ++q) {
    int c = tid * 3 + q;
    float o = (y[q] - mu) * rstd * gamma[c] + beta[c];
    xr[c] = o;
    if (outp) outp[((size_t)b * TT + t) * HH + c] = o;
  }
}

// ---------------------------------------------------------------- label/image dst nodes (data-dependent fan-in)
__global__ void agg_special_kernel(const float* __restrict__ Hmat, const float* __restrict__ ssrc,
                                   const float* __restrict__ sdst, const int* __restrict__ idxl,
                                   const int* __restrict__ idxi, const float* __restrict__ bias,
                                   const float* __restrict__ gamma, const float* __restrict__ beta,
                                   float* __restrict__ x) {
  const int j = blockIdx.x;            // 0..7 -> node 512+j
  const int b = blockIdx.y;
  const int tid = threadIdx.x;         // 256
  const int n = TT + j;
  const bool isLabel = (j < 4);
  const int target = isLabel ? j : (j - 4);

  __shared__ float se[NNODE * NHEAD];  // per-candidate, per-head e (NEG_INF = invalid)
  __shared__ float red[4 * 256];
  __shared__ float smax[NHEAD], sden[NHEAD];

  float sd[NHEAD];
#pragma unroll
  for (int hd = 0; hd < NHEAD; ++hd) sd[hd] = sdst[((size_t)b * NNODE + n) * NHEAD + hd];

  // phase 1: edge validity + raw scores into LDS
  for (int c = tid; c < NNODE; c += 256) {
    bool valid;
    if (c < TT) {
      const int* ix = (isLabel ? idxl : idxi) + ((size_t)b * TT + c) * 3;
      valid = (ix[0] == target) | (ix[1] == target) | (ix[2] == target);
    } else {
      valid = true;                    // all label/image<->label/image + self edges exist for these dsts
    }
#pragma unroll
    for (int hd = 0; hd < NHEAD; ++hd) {
      float v = NEG_INF;
      if (valid) {
        v = ssrc[((size_t)b * NNODE + c) * NHEAD + hd] + sd[hd];
        v = v > 0.f ? v : 0.2f * v;
      }
      se[c * NHEAD + hd] = v;
    }
  }
  __syncthreads();

  // phase 2a: per-head max
  float pm[NHEAD] = {NEG_INF, NEG_INF, NEG_INF, NEG_INF};
  for (int c = tid; c < NNODE; c += 256)
#pragma unroll
    for (int hd = 0; hd < NHEAD; ++hd) pm[hd] = fmaxf(pm[hd], se[c * NHEAD + hd]);
#pragma unroll
  for (int hd = 0; hd < NHEAD; ++hd) red[hd * 256 + tid] = pm[hd];
  __syncthreads();
  for (int s = 128; s > 0; s >>= 1) {
    if (tid < s)
#pragma unroll
      for (int hd = 0; hd < NHEAD; ++hd)
        red[hd * 256 + tid] = fmaxf(red[hd * 256 + tid], red[hd * 256 + tid + s]);
    __syncthreads();
  }
  if (tid < NHEAD) smax[tid] = red[tid * 256];
  __syncthreads();

  // phase 2b: exp in place + per-head denominator
  float pd2[NHEAD] = {0.f, 0.f, 0.f, 0.f};
  for (int c = tid; c < NNODE; c += 256)
#pragma unroll
    for (int hd = 0; hd < NHEAD; ++hd) {
      float v = se[c * NHEAD + hd];
      v = (v == NEG_INF) ? 0.f : __expf(v - smax[hd]);
      se[c * NHEAD + hd] = v;
      pd2[hd] += v;
    }
#pragma unroll
  for (int hd = 0; hd < NHEAD; ++hd) red[hd * 256 + tid] = pd2[hd];
  __syncthreads();
  for (int s = 128; s > 0; s >>= 1) {
    if (tid < s)
#pragma unroll
      for (int hd = 0; hd < NHEAD; ++hd) red[hd * 256 + tid] += red[hd * 256 + tid + s];
    __syncthreads();
  }
  if (tid < NHEAD) sden[tid] = red[tid * 256];
  __syncthreads();

  // phase 3: weighted gather of h[src] per channel
  const int hd = tid >> 6;
  const float inv = 1.0f / (sden[hd] + 1e-16f);
  float acc[3] = {0.f, 0.f, 0.f};
  for (int c = 0; c < NNODE; ++c) {
    float w = se[c * NHEAD + hd];
    if (w != 0.f) {
      w *= inv;
      const float* hr = Hmat + ((size_t)b * NNODE + c) * HH;
#pragma unroll
      for (int q = 0; q < 3; ++q) acc[q] += w * hr[tid * 3 + q];
    }
  }

  // epilogue: ReLU + residual + LN (reuse red[] for the row reduction)
  float* xr = x + ((size_t)b * NNODE + n) * HH;
  float y[3]; float s1 = 0.f, s2 = 0.f;
#pragma unroll
  for (int q = 0; q < 3; ++q) {
    int c = tid * 3 + q;
    float o = fmaxf(acc[q] + bias[c], 0.f);
    float yy = o + xr[c];
    y[q] = yy; s1 += yy; s2 += yy * yy;
  }
  __syncthreads();
  red[tid] = s1; red[256 + tid] = s2; __syncthreads();
  for (int s = 128; s > 0; s >>= 1) {
    if (tid < s) { red[tid] += red[tid + s]; red[256 + tid] += red[256 + tid + s]; }
    __syncthreads();
  }
  const float mu = red[0] * (1.0f / HH);
  const float var = red[256] * (1.0f / HH) - mu * mu;
  const float rstd = rsqrtf(var + LN_EPS);
#pragma unroll
  for (int q = 0; q < 3; ++q) {
    int c = tid * 3 + q;
    xr[c] = (y[q] - mu) * rstd * gamma[c] + beta[c];
  }
}

// ----------------------------------------------------------------
extern "C" void kernel_launch(void* const* d_in, const int* in_sizes, int n_in,
                              void* d_out, int out_size, void* d_ws, size_t ws_size,
                              hipStream_t stream) {
  (void)in_sizes; (void)n_in; (void)out_size; (void)ws_size;
  const float* text   = (const float*)d_in[0];
  const float* label  = (const float*)d_in[1];
  const float* image  = (const float*)d_in[2];
  const float* Ws     = (const float*)d_in[3];
  const float* a_srcs = (const float*)d_in[4];
  const float* a_dsts = (const float*)d_in[5];
  const float* biases = (const float*)d_in[6];
  const float* gammas = (const float*)d_in[7];
  const float* betas  = (const float*)d_in[8];
  float* out = (float*)d_out;

  char* ws = (char*)d_ws;
  size_t off = 0;
  auto alloc = [&](size_t nbytes) -> void* {
    void* p = ws + off;
    off += (nbytes + 255) & ~(size_t)255;
    return p;
  };
  float* x           = (float*)alloc((size_t)BATCH * NNODE * HH * 4);
  float* hbuf        = (float*)alloc((size_t)BATCH * NNODE * HH * 4);
  unsigned int* Wp   = (unsigned int*)alloc((size_t)NLAYER * 48 * 24 * 256 * 4);
  unsigned int* Apck = (unsigned int*)alloc((size_t)BATCH * MT * 24 * 256 * 4);
  float* ssrc        = (float*)alloc((size_t)BATCH * NNODE * NHEAD * 4);
  float* sdst        = (float*)alloc((size_t)BATCH * NNODE * NHEAD * 4);
  int*   idxl        = (int*)alloc((size_t)BATCH * TT * 3 * 4);
  int*   idxi        = (int*)alloc((size_t)BATCH * TT * 3 * 4);

  const long total = (long)BATCH * NNODE * HH;
  build_x_kernel<<<dim3((unsigned)((total + 255) / 256)), dim3(256), 0, stream>>>(
      text, label, image, x, total);
  pack_w_kernel<<<dim3(24, 48, NLAYER), dim3(32), 0, stream>>>(Ws, Wp);
  topk_kernel<<<dim3(TT, BATCH), dim3(128), 0, stream>>>(text, label, image, idxl, idxi);

  for (int l = 0; l < NLAYER; ++l) {
    pack_a_kernel<<<dim3(24, MT, BATCH), dim3(32), 0, stream>>>(x, Apck);
    gemm_wmma_kernel<<<dim3(12, MT, BATCH), dim3(32), 0, stream>>>(
        Apck, Wp + (size_t)l * 48 * 24 * 256, hbuf);
    scores_kernel<<<dim3(NNODE, BATCH), dim3(256), 0, stream>>>(
        hbuf, a_srcs + (size_t)l * HH, a_dsts + (size_t)l * HH, ssrc, sdst);
    agg_text_kernel<<<dim3(TT, BATCH), dim3(256), 0, stream>>>(
        hbuf, ssrc, sdst, idxl, idxi,
        biases + (size_t)l * HH, gammas + (size_t)l * HH, betas + (size_t)l * HH,
        x, (l == NLAYER - 1) ? out : nullptr);
    if (l < NLAYER - 1) {  // label/image rows are dead after the last layer
      agg_special_kernel<<<dim3(8, BATCH), dim3(256), 0, stream>>>(
          hbuf, ssrc, sdst, idxl, idxi,
          biases + (size_t)l * HH, gammas + (size_t)l * HH, betas + (size_t)l * HH, x);
    }
  }
}
